// GINENet_30124900614321
// MI455X (gfx1250) — compile-verified, matrix operand order checked by reading, not run on hardware
//
#include <hip/hip_runtime.h>
#include <hip/hip_bf16.h>

// ---------------- problem constants (fixed by the reference) ----------------
#define N_NODES   51200
#define N_EDGES   819200
#define D_IN      128
#define D_H1      256          // 2*D_IN
#define D_HID     256
#define N_GRAPHS  256
#define NODES_PER_GRAPH (N_NODES / N_GRAPHS)   // 200
#define D_OUT     16
#define TILE_M    32           // node rows per workgroup in the fused MLP

// ---------------- vector types ----------------
typedef __attribute__((ext_vector_type(16))) __bf16       v16bf;
typedef __attribute__((ext_vector_type(8)))  float        v8f;
typedef __attribute__((ext_vector_type(4)))  unsigned int u32x4;
typedef __attribute__((ext_vector_type(8)))  int          i32x8;
typedef __attribute__((ext_vector_type(4)))  int          i32x4;

#ifndef __has_builtin
#define __has_builtin(x) 0
#endif
#if __has_builtin(__builtin_amdgcn_tensor_load_to_lds) && __has_builtin(__builtin_amdgcn_s_wait_tensorcnt)
#define USE_TDM 1
#else
#define USE_TDM 0
#endif

// ---------------- helpers ----------------
__device__ __forceinline__ unsigned short bf_bits(float f) {
  union { float f; unsigned u; } v; v.f = f;
  unsigned r = v.u + 0x7FFFu + ((v.u >> 16) & 1u);   // round-to-nearest-even
  return (unsigned short)(r >> 16);
}
__device__ __forceinline__ __bf16 f2bf(float f) {
  union { unsigned short s; __bf16 b; } u; u.s = bf_bits(f); return u.b;
}
__device__ __forceinline__ __bf16 bits2bf(unsigned short h) {
  union { unsigned short s; __bf16 b; } u; u.s = h; return u.b;
}
// K index of element j (0..15) for 16-bit A/B WMMA fragments (ISA 7.12.2),
// hi = lane>>4.  VGPR v holds K pairs: (v<4?0:16) + hi*8 + (v&3)*2 + parity.
__device__ __forceinline__ int frag_k(int j, int hi) {
  const int v = j >> 1;
  return ((v & 4) << 2) | (hi << 3) | ((v & 3) << 1) | (j & 1);
}

// ================= kernel 1: aggr = (1+eps)*x =================
__global__ __launch_bounds__(256) void gine_init_aggr(const float* __restrict__ x,
                                                      const float* __restrict__ eps,
                                                      float* __restrict__ aggr) {
  const float s = 1.0f + eps[0];
  const size_t i = (size_t)blockIdx.x * 256 + threadIdx.x;  // float4 index
  float4 v = ((const float4*)x)[i];
  v.x *= s; v.y *= s; v.z *= s; v.w *= s;
  ((float4*)aggr)[i] = v;
}

// ================= kernel 2: pack weights to fragment-ordered bf16 =========
// w1: [128 x 256] -> 4x16 fragments;  w2: [256 x 256] -> 8x16 fragments.
// Fragment (kt,nt) stored as 512 contiguous bf16: lane-major, 16 elems/lane.
__global__ __launch_bounds__(256) void gine_pack_weights(
    const float* __restrict__ w1, const float* __restrict__ w2,
    __bf16* __restrict__ w1p, __bf16* __restrict__ w2p)
{
  const int gid  = blockIdx.x * 256 + threadIdx.x;
  const int r    = gid & 511;
  const int lane = r >> 4;
  const int j    = r & 15;
  const int hi   = lane >> 4;
  const int nlo  = lane & 15;
  if (gid < 4 * 16 * 512) {                       // w1 region
    const int frag = gid >> 9;
    const int kt = frag >> 4, nt = frag & 15;
    const int K = kt * 32 + frag_k(j, hi);
    const int N = nt * 16 + nlo;
    w1p[gid] = f2bf(w1[K * D_H1 + N]);
  } else {                                        // w2 region
    const int g = gid - 4 * 16 * 512;
    const int frag = g >> 9;
    const int kt = frag >> 4, nt = frag & 15;
    const int K = kt * 32 + frag_k(j, hi);
    const int N = nt * 16 + nlo;
    w2p[g] = f2bf(w2[K * D_HID + N]);
  }
}

// ================= kernel 3: edge message + scatter-sum =================
// thread t: edge e = t>>5, feature chunk f = (t&31)*4
__global__ __launch_bounds__(256) void gine_edge_scatter(
    const float* __restrict__ x, const int* __restrict__ ei,
    const float* __restrict__ ew, float* __restrict__ aggr)
{
  const size_t t = (size_t)blockIdx.x * 256 + threadIdx.x;
  const int e = (int)(t >> 5);
  const int f = ((int)t & 31) * 4;
  const int src = ei[e];
  const int dst = ei[N_EDGES + e];
  const float4 xv = *(const float4*)(x  + (size_t)src * D_IN + f);
  const float4 wv = *(const float4*)(ew + (size_t)e   * D_IN + f);
  __builtin_prefetch(ew + (size_t)(e + 16) * D_IN + f, 0, 1);  // global_prefetch
  float4 m;
  m.x = fmaxf(xv.x + wv.x, 0.0f);
  m.y = fmaxf(xv.y + wv.y, 0.0f);
  m.z = fmaxf(xv.z + wv.z, 0.0f);
  m.w = fmaxf(xv.w + wv.w, 0.0f);
  float* o = aggr + (size_t)dst * D_IN + f;
  atomicAdd(o + 0, m.x);
  atomicAdd(o + 1, m.y);
  atomicAdd(o + 2, m.z);
  atomicAdd(o + 3, m.w);
}

// ================= kernel 4: fused MLP (WMMA bf16, f32 accum) =============
// block = 256 threads = 8 waves; each block does 32 node rows.
// wave w owns output columns [w*32, w*32+32) of both layers.
__global__ __launch_bounds__(256) void gine_mlp_fused(
    const float* __restrict__ aggr,
    const __bf16* __restrict__ w1p, const float* __restrict__ b1,
    const __bf16* __restrict__ w2p, const float* __restrict__ b2,
    float* __restrict__ hbuf)
{
  __shared__ float          sA[TILE_M * D_IN];   // 16 KB, f32 input tile
  __shared__ unsigned short sH[TILE_M * D_H1];   // 16 KB, bf16 hidden tile

  const int tid  = threadIdx.x;
  const int lane = tid & 31;
  const int wave = tid >> 5;
  const int row0 = blockIdx.x * TILE_M;
  const int n    = lane & 15;
  const int hi   = lane >> 4;

  // ---- stage 32x128 f32 input tile into LDS via the Tensor Data Mover ----
#if USE_TDM
  if (wave == 0) {
    const unsigned long long ga =
        (unsigned long long)(const void*)(aggr + (size_t)row0 * D_IN);
    const unsigned lds_off = (unsigned)(unsigned long long)(&sA[0]); // low32 = LDS offset
    u32x4 g0;
    g0[0] = 1u;                                        // count=1, user D#
    g0[1] = lds_off;                                   // lds_addr
    g0[2] = (unsigned)(ga & 0xFFFFFFFFull);            // global_addr[31:0]
    g0[3] = (unsigned)((ga >> 32) & 0x1FFFFFFull) | 0x80000000u; // [56:32] + type=2
    i32x8 g1;
    g1[0] = (int)(2u << 16);                 // data_size = 4B
    g1[1] = (int)((unsigned)D_IN << 16);     // tensor_dim0 = 128
    g1[2] = (int)((unsigned)N_NODES << 16);  // tensor_dim1[15:0]
    g1[3] = (int)((unsigned)D_IN << 16);     // tile_dim0 = 128
    g1[4] = TILE_M;                          // tile_dim1 = 32, tile_dim2 = 0
    g1[5] = D_IN;                            // tensor_dim0_stride = 128
    g1[6] = 0;
    g1[7] = 0;
    i32x4 gz;  gz[0] = gz[1] = gz[2] = gz[3] = 0;      // groups 2/3: unused (2D tile)
    i32x8 gz8; gz8[0] = gz8[1] = gz8[2] = gz8[3] = 0;
               gz8[4] = gz8[5] = gz8[6] = gz8[7] = 0;
    __builtin_amdgcn_tensor_load_to_lds(g0, g1, gz, gz, gz8, 0);
    __builtin_amdgcn_s_wait_tensorcnt(0);
  }
  __syncthreads();
#else
  {
    const float4* src = (const float4*)(aggr + (size_t)row0 * D_IN);
    float4* dst = (float4*)sA;
    for (int i = tid; i < TILE_M * D_IN / 4; i += 256) dst[i] = src[i];
    __syncthreads();
  }
#endif

  const int ntB = wave * 2;       // this wave's first 16-col tile (of 16)
  const v8f vzero = {0.f, 0.f, 0.f, 0.f, 0.f, 0.f, 0.f, 0.f};

  // ---------------- layer 1: h1 = relu(A @ w1 + b1) ----------------
  v8f acc[2][2];
  #pragma unroll
  for (int tm = 0; tm < 2; ++tm)
    #pragma unroll
    for (int tn = 0; tn < 2; ++tn) acc[tm][tn] = vzero;

  #pragma unroll
  for (int kt = 0; kt < D_IN / 32; ++kt) {
    v16bf aF[2];
    #pragma unroll
    for (int tm = 0; tm < 2; ++tm) {
      const int m = tm * 16 + n;
      #pragma unroll
      for (int j = 0; j < 16; ++j)
        aF[tm][j] = f2bf(sA[m * D_IN + kt * 32 + frag_k(j, hi)]);
    }
    #pragma unroll
    for (int tn = 0; tn < 2; ++tn) {
      const v16bf bF =
          *(const v16bf*)(w1p + (size_t)(kt * 16 + ntB + tn) * 512 + lane * 16);
      #pragma unroll
      for (int tm = 0; tm < 2; ++tm)
        acc[tm][tn] = __builtin_amdgcn_wmma_f32_16x16x32_bf16(
            false, aF[tm], false, bF, (short)0, acc[tm][tn], false, false);
    }
  }

  // epilogue 1: +bias, relu, bf16 into LDS for layer 2
  #pragma unroll
  for (int tn = 0; tn < 2; ++tn) {
    const int col = (ntB + tn) * 16 + n;
    const float bb = b1[col];
    #pragma unroll
    for (int tm = 0; tm < 2; ++tm) {
      #pragma unroll
      for (int r = 0; r < 8; ++r) {
        const int m = tm * 16 + hi * 8 + r;
        const float v = acc[tm][tn][r] + bb;
        sH[m * D_H1 + col] = bf_bits(v > 0.0f ? v : 0.0f);
      }
    }
  }
  __syncthreads();

  // ---------------- layer 2: h2 = h1 @ w2 + b2 ----------------
  v8f acc2[2][2];
  #pragma unroll
  for (int tm = 0; tm < 2; ++tm)
    #pragma unroll
    for (int tn = 0; tn < 2; ++tn) acc2[tm][tn] = vzero;

  #pragma unroll
  for (int kt = 0; kt < D_H1 / 32; ++kt) {
    v16bf aF[2];
    #pragma unroll
    for (int tm = 0; tm < 2; ++tm) {
      const int m = tm * 16 + n;
      #pragma unroll
      for (int j = 0; j < 16; ++j)
        aF[tm][j] = bits2bf(sH[m * D_H1 + kt * 32 + frag_k(j, hi)]);
    }
    #pragma unroll
    for (int tn = 0; tn < 2; ++tn) {
      const v16bf bF =
          *(const v16bf*)(w2p + (size_t)(kt * 16 + ntB + tn) * 512 + lane * 16);
      #pragma unroll
      for (int tm = 0; tm < 2; ++tm)
        acc2[tm][tn] = __builtin_amdgcn_wmma_f32_16x16x32_bf16(
            false, aF[tm], false, bF, (short)0, acc2[tm][tn], false, false);
    }
  }

  // epilogue 2: +bias, store f32 node features for pooling
  #pragma unroll
  for (int tn = 0; tn < 2; ++tn) {
    const int col = (ntB + tn) * 16 + n;
    const float bb = b2[col];
    #pragma unroll
    for (int tm = 0; tm < 2; ++tm) {
      #pragma unroll
      for (int r = 0; r < 8; ++r) {
        const int row = row0 + tm * 16 + hi * 8 + r;
        hbuf[(size_t)row * D_HID + col] = acc2[tm][tn][r] + bb;
      }
    }
  }
}

// ================= kernel 5: per-graph max pooling =================
// batch is jnp.repeat(arange(256), 200): graph g owns rows [g*200,(g+1)*200)
__global__ __launch_bounds__(256) void gine_pool_max(
    const float* __restrict__ hbuf, float* __restrict__ pooled)
{
  const int g = blockIdx.x, c = threadIdx.x;
  const float* p = hbuf + (size_t)g * NODES_PER_GRAPH * D_HID + c;
  float m = -3.402823466e38f;
  for (int i = 0; i < NODES_PER_GRAPH; ++i)
    m = fmaxf(m, p[(size_t)i * D_HID]);
  pooled[g * D_HID + c] = m;
}

// ================= kernel 6: final linear 256x256 @ 256x16 =================
__global__ __launch_bounds__(256) void gine_final(
    const float* __restrict__ pooled, const float* __restrict__ lw,
    const float* __restrict__ lb, float* __restrict__ out)
{
  const int t = blockIdx.x * 256 + threadIdx.x;   // 4096 outputs
  const int g = t >> 4, o = t & 15;
  const float* p = pooled + g * D_HID;
  float acc = lb[o];
  #pragma unroll 8
  for (int k = 0; k < D_HID; ++k) acc += p[k] * lw[k * D_OUT + o];
  out[t] = acc;
}

// ==========================================================================
extern "C" void kernel_launch(void* const* d_in, const int* in_sizes, int n_in,
                              void* d_out, int out_size, void* d_ws, size_t ws_size,
                              hipStream_t stream) {
  const float* x   = (const float*)d_in[0];
  const int*   ei  = (const int*)  d_in[1];   // [2, E]: row0 = src, row1 = dst
  const float* ew  = (const float*)d_in[2];
  /* d_in[3] = batch: layout is fixed (200 nodes per graph), unused */
  const float* eps = (const float*)d_in[4];
  const float* w1  = (const float*)d_in[5];
  const float* b1  = (const float*)d_in[6];
  const float* w2  = (const float*)d_in[7];
  const float* b2  = (const float*)d_in[8];
  const float* lw  = (const float*)d_in[9];
  const float* lb  = (const float*)d_in[10];
  float* out = (float*)d_out;

  // workspace layout (all offsets 256-B aligned)
  char* ws = (char*)d_ws;
  float*  aggr   = (float*)(ws);                  // 51200*128 f32 = 26,214,400 B
  float*  hbuf   = (float*)(ws + 26214400);       // 51200*256 f32 = 52,428,800 B
  float*  pooled = (float*)(ws + 78643200);       // 256*256   f32 =    262,144 B
  __bf16* w1p    = (__bf16*)(ws + 78905344);      // 128*256  bf16 =     65,536 B
  __bf16* w2p    = (__bf16*)(ws + 78970880);      // 256*256  bf16 =    131,072 B

  gine_init_aggr  <<<(N_NODES * D_IN / 4) / 256, 256, 0, stream>>>(x, eps, aggr);
  gine_pack_weights<<<(4 * 16 * 512 + 8 * 16 * 512) / 256, 256, 0, stream>>>(w1, w2, w1p, w2p);
  gine_edge_scatter<<<(unsigned)((size_t)N_EDGES * 32 / 256), 256, 0, stream>>>(x, ei, ew, aggr);
  gine_mlp_fused  <<<N_NODES / TILE_M, 256, 0, stream>>>(aggr, w1p, b1, w2p, b2, hbuf);
  gine_pool_max   <<<N_GRAPHS, 256, 0, stream>>>(hbuf, pooled);
  gine_final      <<<(N_GRAPHS * D_OUT) / 256, 256, 0, stream>>>(pooled, lw, lb, out);
}
